// WuBuNestingLevel_54159537602600
// MI455X (gfx1250) — matrix operand order repaired and checked by reading, not compiled, not run on hardware
//
#include <hip/hip_runtime.h>
#include <hip/hip_bf16.h>

// ---------------------------------------------------------------------------
// Types for CDNA5 WMMA (wave32, v_wmma_f32_16x16x32_bf16)
// ---------------------------------------------------------------------------
typedef __attribute__((ext_vector_type(16))) __bf16 v16bf;
typedef __attribute__((ext_vector_type(8)))  __bf16 v8bf;
typedef __attribute__((ext_vector_type(4)))  __bf16 v4bf;
typedef __attribute__((ext_vector_type(8)))  float  v8f;

// pointer types for the async global->LDS DMA builtin:
//   void __builtin_amdgcn_global_load_async_to_lds_b128(
//       int4 addrspace(1)* src, int4 addrspace(3)* dst, imm offset, imm cpol)
typedef int v4i __attribute__((vector_size(16)));
typedef __attribute__((address_space(1))) v4i as1_v4i;
typedef __attribute__((address_space(3))) v4i as3_v4i;

union FragAB { v16bf v; v8bf h[2]; };

// ---- optional CDNA5 async global->LDS DMA (ASYNCcnt) ----------------------
#if defined(__has_builtin)
# if __has_builtin(__builtin_amdgcn_global_load_async_to_lds_b128)
#  define USE_ASYNC_LDS 1
# endif
#endif
#ifndef USE_ASYNC_LDS
# define USE_ASYNC_LDS 0
#endif

#if defined(__has_builtin)
# if __has_builtin(__builtin_amdgcn_s_wait_asynccnt)
#  define WAIT_ASYNC() __builtin_amdgcn_s_wait_asynccnt(0)
# else
#  define WAIT_ASYNC() asm volatile("s_wait_asynccnt 0x0" ::: "memory")
# endif
#else
# define WAIT_ASYNC() asm volatile("s_wait_asynccnt 0x0" ::: "memory")
#endif

#define ASYNC_G2L_B128(gptr, lptr)                                          \
  __builtin_amdgcn_global_load_async_to_lds_b128(                           \
      (as1_v4i*)(v4i*)(void*)(gptr), (as3_v4i*)(v4i*)(void*)(lptr), 0, 0)

// ---------------------------------------------------------------------------
// Small device helpers
// ---------------------------------------------------------------------------
__device__ __forceinline__ float n2n(float x) {
  // jnp.nan_to_num defaults: nan->0, +inf->FLT_MAX, -inf->-FLT_MAX
  const float FMAX = 3.4028234663852886e38f;
  if (x != x) return 0.0f;
  if (x ==  __builtin_inff()) return  FMAX;
  if (x == -__builtin_inff()) return -FMAX;
  return x;
}

__device__ __forceinline__ float gelu_tanh(float x) {
  // jax.nn.gelu approximate=True
  const float k0 = 0.7978845608028654f;   // sqrt(2/pi)
  const float k1 = 0.044715f;
  float x3 = x * x * x;
  return 0.5f * x * (1.0f + tanhf(k0 * (x + k1 * x3)));
}

__device__ __forceinline__ float softplus_f(float x) {
  return (x > 20.0f) ? x : log1pf(expf(x));
}

__device__ __forceinline__ float wave_sum32(float x) {
#pragma unroll
  for (int off = 16; off > 0; off >>= 1) x += __shfl_xor(x, off, 32);
  return x;
}

// ---------------------------------------------------------------------------
// Kernel: build full 1024x1024 quaternion weight matrix (bf16, [N,K] row-major)
//   v[m,n] = sum_k x[m,k] * Wq[n,k]
// ---------------------------------------------------------------------------
__global__ __launch_bounds__(256) void build_wq(
    const float* __restrict__ rw, const float* __restrict__ iw,
    const float* __restrict__ jw, const float* __restrict__ kw,
    __bf16* __restrict__ Wq)
{
  int idx = blockIdx.x * 256 + threadIdx.x;      // 0 .. 1024*1024-1
  int n  = idx >> 10;       // output index (row of Wq)
  int kc = idx & 1023;      // input index (col of Wq)
  int o  = n >> 2,  co = n & 3;
  int q  = kc >> 2, ci = kc & 3;
  const float* wsrc[4] = { rw, iw, jw, kw };
  const int   srci[4][4] = { {0,1,2,3}, {1,0,3,2}, {2,3,0,1}, {3,2,1,0} };
  const float sgn [4][4] = { {1.f,-1.f,-1.f,-1.f},
                             {1.f, 1.f, 1.f,-1.f},
                             {1.f,-1.f, 1.f, 1.f},
                             {1.f, 1.f,-1.f, 1.f} };
  float v = sgn[co][ci] * wsrc[srci[co][ci]][o * 256 + q];
  Wq[idx] = (__bf16)v;
}

__global__ __launch_bounds__(256) void cvt_bf16(
    const float* __restrict__ in, __bf16* __restrict__ out, int n)
{
  int i = blockIdx.x * 256 + threadIdx.x;
  if (i < n) out[i] = (__bf16)in[i];
}

// ---------------------------------------------------------------------------
// WMMA GEMM:  Out[M,N] = epilogue( A[M,K] * B[N,K]^T + bias[N] )
//   EPI 0: f32 out                         (quat-linear -> v)
//   EPI 1: gelu, bf16 out                  (flow hidden -> h)
//   EPI 2: nan_to_num + residual, f32 out  (flow out + v -> u)
// 256 thr (8 waves), tile 128x128, BK=32, double-buffered LDS; each wave owns
// a 32x64 patch = 2x4 grid of 16x16 WMMA accumulators.
// ---------------------------------------------------------------------------
template<int EPI, typename AT>
__global__ __launch_bounds__(256) void gemm_wmma(
    const AT* __restrict__ A, int lda,
    const __bf16* __restrict__ B,          // [N,K] row-major bf16
    const float* __restrict__ bias,        // [N]
    const float* __restrict__ Cres,        // [M,N] residual (EPI==2)
    void* __restrict__ Out,
    int M, int N, int K)
{
  __shared__ __bf16 As[2][128 * 32];
  __shared__ __bf16 Bs[2][128 * 32];

  const int tid  = threadIdx.x;
  const int lane = tid & 31;
  const int wid  = tid >> 5;
  const int wm   = wid & 3;     // 4 M-slices of 32 rows
  const int wn   = wid >> 2;    // 2 N-slices of 64 cols
  const int hsel = lane >> 4;   // half-wave select
  const int l16  = lane & 15;
  const int m0 = blockIdx.y * 128;
  const int n0 = blockIdx.x * 128;

  // per-thread staging coordinates
  const int ar4 = tid >> 3, ac4 = (tid & 7) * 4;  // f32 A path: 4 rows x float4
  const int r8  = tid >> 2, c8  = (tid & 3) * 8;  // 16B bf16 path: 2 rows x v8bf

  v8f acc[2][4];
#pragma unroll
  for (int a = 0; a < 2; ++a)
#pragma unroll
    for (int b = 0; b < 4; ++b)
      acc[a][b] = (v8f){0.f,0.f,0.f,0.f,0.f,0.f,0.f,0.f};

  const int nK = K / 32;

  // ---------------- prologue: stage tile 0 into buffer 0 ----------------
  {
    if (sizeof(AT) == 4) {
      const float* Af = (const float*)A;
      float4 areg[4];
#pragma unroll
      for (int it = 0; it < 4; ++it)
        areg[it] = *(const float4*)(Af + (size_t)(m0 + ar4 + it * 32) * lda + ac4);
#pragma unroll
      for (int it = 0; it < 4; ++it) {
        v4bf hv;
        hv[0] = (__bf16)areg[it].x; hv[1] = (__bf16)areg[it].y;
        hv[2] = (__bf16)areg[it].z; hv[3] = (__bf16)areg[it].w;
        *(v4bf*)&As[0][(ar4 + it * 32) * 32 + ac4] = hv;
      }
    } else {
#if USE_ASYNC_LDS
      const __bf16* Ah = (const __bf16*)A;
#pragma unroll
      for (int it = 0; it < 2; ++it)
        ASYNC_G2L_B128(Ah + (size_t)(m0 + r8 + it * 64) * lda + c8,
                       &As[0][(r8 + it * 64) * 32 + c8]);
#else
      const __bf16* Ah = (const __bf16*)A;
#pragma unroll
      for (int it = 0; it < 2; ++it)
        *(v8bf*)&As[0][(r8 + it * 64) * 32 + c8] =
            *(const v8bf*)(Ah + (size_t)(m0 + r8 + it * 64) * lda + c8);
#endif
    }
#if USE_ASYNC_LDS
#pragma unroll
    for (int it = 0; it < 2; ++it)
      ASYNC_G2L_B128(B + (size_t)(n0 + r8 + it * 64) * K + c8,
                     &Bs[0][(r8 + it * 64) * 32 + c8]);
    WAIT_ASYNC();
#else
#pragma unroll
    for (int it = 0; it < 2; ++it)
      *(v8bf*)&Bs[0][(r8 + it * 64) * 32 + c8] =
          *(const v8bf*)(B + (size_t)(n0 + r8 + it * 64) * K + c8);
#endif
    __syncthreads();
  }

  // ---------------- main loop: compute buf[cur], stage buf[nxt] ----------
  for (int kt = 0; kt < nK; ++kt) {
    const int cur = kt & 1;
    const int nxt = cur ^ 1;
    const int kn  = (kt + 1) * 32;          // next tile's K offset
    const bool has_next = (kt + 1 < nK);

    // ---- issue next-tile staging (early, overlaps with WMMA below) ----
    float4 areg[4];
    if (has_next) {
      if (sizeof(AT) == 4) {
        const float* Af = (const float*)A;
#pragma unroll
        for (int it = 0; it < 4; ++it)
          areg[it] = *(const float4*)(Af + (size_t)(m0 + ar4 + it * 32) * lda + kn + ac4);
      } else {
#if USE_ASYNC_LDS
        const __bf16* Ah = (const __bf16*)A;
#pragma unroll
        for (int it = 0; it < 2; ++it)
          ASYNC_G2L_B128(Ah + (size_t)(m0 + r8 + it * 64) * lda + kn + c8,
                         &As[nxt][(r8 + it * 64) * 32 + c8]);
#endif
      }
#if USE_ASYNC_LDS
#pragma unroll
      for (int it = 0; it < 2; ++it)
        ASYNC_G2L_B128(B + (size_t)(n0 + r8 + it * 64) * K + kn + c8,
                       &Bs[nxt][(r8 + it * 64) * 32 + c8]);
#endif
      // prefetch two tiles ahead on the f32 A stream (global_prefetch_b8)
      if (sizeof(AT) == 4 && kt + 2 < nK)
        __builtin_prefetch(
            (const char*)A + ((size_t)(m0 + ar4) * lda + kn + 32) * sizeof(AT), 0, 1);
    }

    // ---- fragments per ISA 7.12.2 bf16 layouts, from buf[cur] ----
    FragAB af[2], bfr[4];
#pragma unroll
    for (int tm = 0; tm < 2; ++tm) {
      int r = wm * 32 + tm * 16 + l16;            // M row
      af[tm].h[0] = *(const v8bf*)&As[cur][r * 32 + hsel * 8];       // K 0-7 / 8-15
      af[tm].h[1] = *(const v8bf*)&As[cur][r * 32 + hsel * 8 + 16];  // K 16-23 / 24-31
    }
#pragma unroll
    for (int tn = 0; tn < 4; ++tn) {
      int r = wn * 64 + tn * 16 + l16;            // N col (row of B)
      bfr[tn].h[0] = *(const v8bf*)&Bs[cur][r * 32 + hsel * 8];      // K 0-7 / 8-15
      bfr[tn].h[1] = *(const v8bf*)&Bs[cur][r * 32 + hsel * 8 + 16]; // K 16-23 / 24-31
    }
#pragma unroll
    for (int tm = 0; tm < 2; ++tm)
#pragma unroll
      for (int tn = 0; tn < 4; ++tn)
        acc[tm][tn] = __builtin_amdgcn_wmma_f32_16x16x32_bf16(
            false, af[tm].v, false, bfr[tn].v,
            (short)0, acc[tm][tn], false, false);

    // ---- finish staging next tile ----
    if (has_next) {
      if (sizeof(AT) == 4) {
#pragma unroll
        for (int it = 0; it < 4; ++it) {
          v4bf hv;
          hv[0] = (__bf16)areg[it].x; hv[1] = (__bf16)areg[it].y;
          hv[2] = (__bf16)areg[it].z; hv[3] = (__bf16)areg[it].w;
          *(v4bf*)&As[nxt][(ar4 + it * 32) * 32 + ac4] = hv;
        }
      }
#if USE_ASYNC_LDS
      WAIT_ASYNC();
#else
      if (sizeof(AT) == 2) {
        const __bf16* Ah = (const __bf16*)A;
#pragma unroll
        for (int it = 0; it < 2; ++it)
          *(v8bf*)&As[nxt][(r8 + it * 64) * 32 + c8] =
              *(const v8bf*)(Ah + (size_t)(m0 + r8 + it * 64) * lda + kn + c8);
      }
#pragma unroll
      for (int it = 0; it < 2; ++it)
        *(v8bf*)&Bs[nxt][(r8 + it * 64) * 32 + c8] =
            *(const v8bf*)(B + (size_t)(n0 + r8 + it * 64) * K + kn + c8);
#endif
    }
    __syncthreads();
  }

  // ---- epilogue: C/D layout — lane l, vgpr v -> (m = v + 8*(l/16), n = l%16)
#pragma unroll
  for (int tm = 0; tm < 2; ++tm) {
#pragma unroll
    for (int tn = 0; tn < 4; ++tn) {
      int gm = m0 + wm * 32 + tm * 16 + hsel * 8;
      int gn = n0 + wn * 64 + tn * 16 + l16;
      float bv = bias[gn];
#pragma unroll
      for (int vg = 0; vg < 8; ++vg) {
        size_t idx = (size_t)(gm + vg) * N + gn;
        float val = acc[tm][tn][vg] + bv;
        if (EPI == 0) {
          ((float*)Out)[idx] = val;
        } else if (EPI == 1) {
          ((__bf16*)Out)[idx] = (__bf16)gelu_tanh(val);
        } else {
          ((float*)Out)[idx] = n2n(val) + Cres[idx];
        }
      }
    }
  }
}

// ---------------------------------------------------------------------------
// Final elementwise kernel: exp-map + log-map on Poincare ball.
// One wave32 per token (D=1024 -> 32 f32/lane), in-place on d_out.
// ---------------------------------------------------------------------------
__global__ __launch_bounds__(256) void hyper_maps(
    float* __restrict__ U,
    const float* __restrict__ plc, const float* __restrict__ pls)
{
  const int lane = threadIdx.x & 31;
  const int tok  = blockIdx.x * 8 + (threadIdx.x >> 5);
  float* row = U + (size_t)tok * 1024;

  const float c = softplus_f(plc[0]) + 1e-7f;
  const float s = softplus_f(pls[0]) + 1e-7f;

  float u[32];
#pragma unroll
  for (int p = 0; p < 8; ++p) {
    float4 f = *(const float4*)(row + p * 128 + lane * 4);
    u[p*4+0] = f.x; u[p*4+1] = f.y; u[p*4+2] = f.z; u[p*4+3] = f.w;
  }

  // ---- exponential_map(v*s, c), eps=1e-8 ----
#pragma unroll
  for (int i = 0; i < 32; ++i) u[i] *= s;
  float ss = 0.f;
#pragma unroll
  for (int i = 0; i < 32; ++i) ss += u[i] * u[i];
  ss = wave_sum32(ss);
  float vnorm = sqrtf(fmaxf(ss, 0.f) + 1e-8f);
  float sc8   = sqrtf(fmaxf(c, 1e-8f));
  float lam   = (vnorm > 1e-8f) ? (tanhf(sc8 * vnorm) / (sc8 * vnorm + 1e-8f))
                                : 1.0f;
#pragma unroll
  for (int i = 0; i < 32; ++i) u[i] *= lam;

  // ---- poincare_clip #1 (eps=1e-5) ----
  float sc5  = sqrtf(fmaxf(c, 1e-5f));
  float maxn = (1.0f / sc5) * (1.0f - 1e-5f);
  ss = 0.f;
#pragma unroll
  for (int i = 0; i < 32; ++i) ss += u[i] * u[i];
  ss = wave_sum32(ss);
  float n0  = sqrtf(fmaxf(ss, 0.f) + 1e-5f);
  float scl = (n0 > maxn) ? (maxn / (n0 + 1e-5f)) : 1.0f;
#pragma unroll
  for (int i = 0; i < 32; ++i) u[i] = n2n(u[i] * scl);

  // ---- logarithmic_map: clip #2 ----
  ss = 0.f;
#pragma unroll
  for (int i = 0; i < 32; ++i) ss += u[i] * u[i];
  ss = wave_sum32(ss);
  float n1   = sqrtf(fmaxf(ss, 0.f) + 1e-5f);
  float scl1 = (n1 > maxn) ? (maxn / (n1 + 1e-5f)) : 1.0f;
#pragma unroll
  for (int i = 0; i < 32; ++i) u[i] = n2n(u[i] * scl1);

  // ---- atanh scaling (eps=1e-7) ----
  ss = 0.f;
#pragma unroll
  for (int i = 0; i < 32; ++i) ss += u[i] * u[i];
  ss = wave_sum32(ss);
  float ynorm = sqrtf(fmaxf(ss, 0.f) + 1e-7f);
  float sc7   = sqrtf(fmaxf(c, 1e-7f));
  float at    = fminf(fmaxf(sc7 * ynorm, -1.0f + 1e-7f), 1.0f - 1e-7f);
  float lam2  = (ynorm > 1e-7f) ? (atanhf(at) / (sc7 * ynorm + 1e-7f)) : 1.0f;
  float inv_s = 1.0f / s;
#pragma unroll
  for (int i = 0; i < 32; ++i) u[i] = n2n(lam2 * u[i]) * inv_s;

#pragma unroll
  for (int p = 0; p < 8; ++p) {
    float4 f;
    f.x = u[p*4+0]; f.y = u[p*4+1]; f.z = u[p*4+2]; f.w = u[p*4+3];
    *(float4*)(row + p * 128 + lane * 4) = f;
  }
}

// ---------------------------------------------------------------------------
// Host launcher
// ---------------------------------------------------------------------------
extern "C" void kernel_launch(void* const* d_in, const int* in_sizes, int n_in,
                              void* d_out, int out_size, void* d_ws, size_t ws_size,
                              hipStream_t stream) {
  (void)in_sizes; (void)n_in; (void)out_size; (void)ws_size;
  const float* x    = (const float*)d_in[0];
  const float* rw   = (const float*)d_in[1];
  const float* iw   = (const float*)d_in[2];
  const float* jw   = (const float*)d_in[3];
  const float* kw   = (const float*)d_in[4];
  const float* bias = (const float*)d_in[5];
  const float* w1   = (const float*)d_in[6];
  const float* b1   = (const float*)d_in[7];
  const float* w2   = (const float*)d_in[8];
  const float* b2   = (const float*)d_in[9];
  const float* lc   = (const float*)d_in[10];
  const float* ls   = (const float*)d_in[11];

  const int M = 8 * 4096;   // 32768 tokens
  const int D = 1024;
  const int H = 512;

  char* wsb = (char*)d_ws;
  __bf16* Wq = (__bf16*)(wsb);                                   // 2 MB
  __bf16* W1 = (__bf16*)(wsb + (size_t)2 * 1024 * 1024);         // 1 MB
  __bf16* W2 = (__bf16*)(wsb + (size_t)3 * 1024 * 1024);         // 1 MB
  float*  V  = (float*) (wsb + (size_t)4 * 1024 * 1024);         // 128 MB
  __bf16* Hb = (__bf16*)(wsb + (size_t)4 * 1024 * 1024
                             + (size_t)M * D * sizeof(float));   // 32 MB

  // Precompute structured weights in bf16
  build_wq<<<(D * D) / 256, 256, 0, stream>>>(rw, iw, jw, kw, Wq);
  cvt_bf16<<<(H * D) / 256, 256, 0, stream>>>(w1, W1, H * D);
  cvt_bf16<<<(D * H) / 256, 256, 0, stream>>>(w2, W2, D * H);

  // v = x @ Wq^T + bias
  gemm_wmma<0, float ><<<dim3(D / 128, M / 128), 256, 0, stream>>>(
      x, D, Wq, bias, nullptr, V, M, D, D);
  // h = gelu(v @ W1^T + b1)  (bf16)
  gemm_wmma<1, float ><<<dim3(H / 128, M / 128), 256, 0, stream>>>(
      V, D, W1, b1, nullptr, Hb, M, H, D);
  // u = nan_to_num(h @ W2^T + b2) + v   -> d_out
  gemm_wmma<2, __bf16><<<dim3(D / 128, M / 128), 256, 0, stream>>>(
      Hb, H, W2, b2, V, d_out, M, D, H);

  // exp-map / log-map, in place on d_out
  hyper_maps<<<M / 8, 256, 0, stream>>>((float*)d_out, lc, ls);
}
